// LayoutLMv3SelfAttention_19473381720541
// MI455X (gfx1250) — matrix operand order repaired and checked by reading, not compiled, not run on hardware
//
#include <hip/hip_runtime.h>
#include <hip/hip_bf16.h>

// ---------------- problem constants ----------------
#define BB 2
#define SS 2048
#define DD 768
#define HH 12
#define DH 64
#define BS (BB*SS)          // 4096 flat rows

typedef __attribute__((ext_vector_type(16))) __bf16 v16bf;
typedef __attribute__((ext_vector_type(8)))  float  v8f;

union V16 { uint4 u[2]; v16bf v; };   // 32 bytes: one WMMA bf16 operand

__device__ __forceinline__ unsigned short f2bf(float x) {
    union { __bf16 h; unsigned short u; } c;
    c.h = (__bf16)x;
    return c.u;
}

// gfx1250 async copy: global -> LDS, 16 bytes per lane, tracked by ASYNCcnt.
__device__ __forceinline__ void async_b128(unsigned lds_byte_off, const void* gsrc) {
    asm volatile("global_load_async_to_lds_b128 %0, %1, off"
                 :: "v"(lds_byte_off),
                    "v"((unsigned long long)(uintptr_t)gsrc)
                 : "memory");
}
__device__ __forceinline__ void wait_async0() {
    asm volatile("s_wait_asynccnt 0" ::: "memory");
}
// low 32 bits of a flat pointer into LDS == LDS byte offset (ISA 10.2 aperture rule)
__device__ __forceinline__ unsigned lds_off32(const void* p) {
    return (unsigned)(uintptr_t)p;
}

// ---------------- kernel 1: fp32 -> bf16 (+ weight transpose) ----------------
__global__ void cvt_kernel(const float* __restrict__ X,
                           const float* __restrict__ Wq,
                           const float* __restrict__ Wk,
                           const float* __restrict__ Wv,
                           unsigned short* __restrict__ Xb,
                           unsigned short* __restrict__ Wtq,
                           unsigned short* __restrict__ Wtk,
                           unsigned short* __restrict__ Wtv) {
    const int i = blockIdx.x * blockDim.x + threadIdx.x;
    const int NX = BS * DD;
    const int NW = DD * DD;
    if (i < NX) Xb[i] = f2bf(X[i]);
    if (i < NW) {
        int k = i / DD, n = i % DD;       // W is [k][n]; store transposed [n][k]
        Wtq[n * DD + k] = f2bf(Wq[i]);
        Wtk[n * DD + k] = f2bf(Wk[i]);
        Wtv[n * DD + k] = f2bf(Wv[i]);
    }
}

// ---------------- kernel 2: QKV projection GEMM (bf16 WMMA) ----------------
// Each wave computes one 16x16 output tile of [4096 x 768] = X @ W + b.
// vT==0: out is [B,H,S,DH] bf16.  vT==1: out is [B,H,DH,S] bf16 (V transposed).
__global__ void qkv_gemm(const unsigned short* __restrict__ Xb,
                         const unsigned short* __restrict__ Wt,   // [n][k] bf16
                         const float* __restrict__ bias,
                         unsigned short* __restrict__ out, int vT) {
    const int lane = threadIdx.x & 31;
    const int wave = threadIdx.x >> 5;
    const int tile = blockIdx.x * 8 + wave;         // 0 .. 256*48-1
    const int mt = tile / 48, nt = tile % 48;
    const int m0 = mt * 16, n0 = nt * 16;
    const int l15 = lane & 15, hihalf = lane >> 4;
    const int akb = hihalf * 8;                     // A layout lo-K base

    const unsigned short* aptr = Xb + (size_t)(m0 + l15) * DD;
    const unsigned short* bptr = Wt + (size_t)(n0 + l15) * DD + hihalf * 8;

    v8f acc = {};
    #pragma unroll 4
    for (int k = 0; k < DD; k += 32) {
        V16 A, Bm;
        // 16-bit A 16x32 layout: lo 8 halves = K kb..kb+7, hi 8 = K 16+kb..16+kb+7
        A.u[0] = *(const uint4*)(aptr + k + akb);
        A.u[1] = *(const uint4*)(aptr + k + akb + 16);
        // 16-bit B 32x16 layout: lane n = l15, lo 8 = K kb..kb+7, hi 8 = K 16+kb..16+kb+7
        Bm.u[0] = *(const uint4*)(bptr + k);
        Bm.u[1] = *(const uint4*)(bptr + k + 16);
        acc = __builtin_amdgcn_wmma_f32_16x16x32_bf16(false, A.v, false, Bm.v,
                                                      (short)0, acc, false, false);
    }

    const float bvv = bias[n0 + l15];
    const int h = n0 / DH, dh0 = n0 % DH;
    #pragma unroll
    for (int r = 0; r < 8; ++r) {
        const int row = m0 + r + hihalf * 8;        // C layout: lanes>=16 hold M=8+r
        const int b = row >> 11, s = row & (SS - 1);
        const unsigned short hv = f2bf(acc[r] + bvv);
        if (!vT)
            out[(((size_t)(b * HH + h) * SS + s) * DH) + dh0 + l15] = hv;
        else
            out[(((size_t)(b * HH + h) * DH + dh0 + l15) * SS) + s] = hv;
    }
}

// ---------------- kernel 3: fused flash attention (bf16 WMMA) ----------------
// Block = 8 waves, all on the same (b,h); wave w owns 16 query rows.
// K/V chunks (64 keys) are async-staged once per block into double-buffered LDS.
#define LDS_KOFF(buf)  ((buf) * 64 * 72)
#define LDS_VOFF(buf)  (2 * 64 * 72 + (buf) * 64 * 72)
#define LDS_POFF(wv)   (4 * 64 * 72 + (wv) * 16 * 72)
#define LDS_HALVES     (4 * 64 * 72 + 8 * 16 * 72)

__global__ void attn_kernel(const unsigned short* __restrict__ Q,   // [B,H,S,DH] bf16
                            const unsigned short* __restrict__ K,   // [B,H,S,DH] bf16
                            const unsigned short* __restrict__ Vt,  // [B,H,DH,S] bf16
                            const float* __restrict__ relp,         // [1,H,S,S]
                            const float* __restrict__ rel2,         // [1,H,S,S]
                            const float* __restrict__ mask,         // [B,1,1,S]
                            float* __restrict__ out) {              // [B,S,D]
    __shared__ __align__(16) unsigned short smem[LDS_HALVES];

    const int tid  = threadIdx.x;
    const int lane = tid & 31;
    const int wave = tid >> 5;
    const int bh = blockIdx.x >> 4;                 // b*H + h
    const int rt = blockIdx.x & 15;                 // 128-row slab within (b,h)
    const int b = bh / HH, h = bh % HH;
    const int l15 = lane & 15, hihalf = lane >> 4;
    const int akb = hihalf * 8;
    const int m0 = rt * 128 + wave * 16;

    const unsigned short* kbase = K  + (size_t)bh * SS * DH;
    const unsigned short* vbase = Vt + (size_t)bh * DH * SS;
    const float* rpb = relp + (size_t)h * SS * SS;
    const float* r2b = rel2 + (size_t)h * SS * SS;
    const float* mkp = mask + (size_t)b * SS;
    unsigned short* myP = &smem[LDS_POFF(wave)];

    // --- Q tile into registers, A layout, both K-chunks (dh 0..31 / 32..63) ---
    const unsigned short* qbase = Q + ((size_t)bh * SS + m0 + l15) * DH;
    V16 qA[2];
    #pragma unroll
    for (int c = 0; c < 2; ++c) {
        qA[c].u[0] = *(const uint4*)(qbase + c * 32 + akb);
        qA[c].u[1] = *(const uint4*)(qbase + c * 32 + akb + 16);
    }

    float mrow[8], lrow[8];
    v8f O[4];
    #pragma unroll
    for (int r = 0; r < 8; ++r) { mrow[r] = -INFINITY; lrow[r] = 0.f; }
    #pragma unroll
    for (int nt = 0; nt < 4; ++nt) O[nt] = (v8f){};

    // async stage of one 64-key chunk (K rows + V^T rows) into LDS buffer `buf`
    auto stage = [&](int buf, int t0) {
        #pragma unroll
        for (int j = 0; j < 2; ++j) {
            const int seg = tid + j * 256;          // 512 16B segments per tensor
            const int row = seg >> 3, c8 = (seg & 7) * 8;
            async_b128(lds_off32(&smem[LDS_KOFF(buf) + row * 72 + c8]),
                       kbase + (size_t)(t0 + row) * DH + c8);
            async_b128(lds_off32(&smem[LDS_VOFF(buf) + row * 72 + c8]),
                       vbase + (size_t)row * SS + t0 + c8);
        }
    };

    stage(0, 0);
    wait_async0();
    __syncthreads();

    for (int t0 = 0; t0 < SS; t0 += 64) {
        const int cur = (t0 >> 6) & 1;
        if (t0 + 64 < SS) stage(cur ^ 1, t0 + 64);   // prefetch next chunk

        // ---- scores S = Q @ K^T for 4 key tiles of 16 (K from LDS) ----
        float sv[4][8];
        #pragma unroll
        for (int nt = 0; nt < 4; ++nt) {
            v8f s = {};
            const unsigned short* kp =
                &smem[LDS_KOFF(cur) + (nt * 16 + l15) * 72 + hihalf * 8];
            #pragma unroll
            for (int c = 0; c < 2; ++c) {
                V16 Bm;
                Bm.u[0] = *(const uint4*)(kp + c * 32);
                Bm.u[1] = *(const uint4*)(kp + c * 32 + 16);
                s = __builtin_amdgcn_wmma_f32_16x16x32_bf16(false, qA[c].v, false, Bm.v,
                                                            (short)0, s, false, false);
            }
            const int key = t0 + nt * 16 + l15;
            const float mval = mkp[key];
            #pragma unroll
            for (int r = 0; r < 8; ++r) {
                const int row = m0 + r + hihalf * 8;
                const size_t off = (size_t)row * SS + key;  // streams rel biases once
                if (nt == 0 && t0 + 256 < SS) {             // prefetch 4 chunks ahead
                    __builtin_prefetch(&rpb[off + 256], 0, 0);
                    __builtin_prefetch(&r2b[off + 256], 0, 0);
                }
                sv[nt][r] = (s[r] + rpb[off] + r2b[off]) * 0.125f + mval;
            }
        }

        // ---- online softmax (PB-relax == max-subtracted softmax exactly) ----
        #pragma unroll
        for (int r = 0; r < 8; ++r) {
            float rm = fmaxf(fmaxf(sv[0][r], sv[1][r]), fmaxf(sv[2][r], sv[3][r]));
            rm = fmaxf(rm, __shfl_xor(rm, 1, 16));
            rm = fmaxf(rm, __shfl_xor(rm, 2, 16));
            rm = fmaxf(rm, __shfl_xor(rm, 4, 16));
            rm = fmaxf(rm, __shfl_xor(rm, 8, 16));
            const float mnew = fmaxf(mrow[r], rm);
            const float sc = __expf(mrow[r] - mnew);    // 0 on first chunk
            mrow[r] = mnew;
            float rs = 0.f;
            #pragma unroll
            for (int nt = 0; nt < 4; ++nt) {
                const float p = __expf(sv[nt][r] - mnew);
                sv[nt][r] = p;
                rs += p;
            }
            rs += __shfl_xor(rs, 1, 16);
            rs += __shfl_xor(rs, 2, 16);
            rs += __shfl_xor(rs, 4, 16);
            rs += __shfl_xor(rs, 8, 16);
            lrow[r] = lrow[r] * sc + rs;
            #pragma unroll
            for (int nt = 0; nt < 4; ++nt) O[nt][r] *= sc;
        }

        // ---- P (C layout) -> LDS -> A layout (same wave; DScnt keeps order) ----
        #pragma unroll
        for (int nt = 0; nt < 4; ++nt)
            #pragma unroll
            for (int r = 0; r < 8; ++r)
                myP[(r + hihalf * 8) * 72 + nt * 16 + l15] = f2bf(sv[nt][r]);

        // ---- O += P @ V  (V^T from LDS: two runs of 8 K-values per lane, table order) ----
        #pragma unroll
        for (int c = 0; c < 2; ++c) {
            V16 pA;
            const unsigned short* pr = myP + l15 * 72 + c * 32 + akb;
            pA.u[0] = *(const uint4*)(pr);
            pA.u[1] = *(const uint4*)(pr + 16);
            #pragma unroll
            for (int nt = 0; nt < 4; ++nt) {
                const unsigned short* vp =
                    &smem[LDS_VOFF(cur) + (nt * 16 + l15) * 72 + c * 32 + hihalf * 8];
                V16 Bm;
                Bm.u[0] = *(const uint4*)(vp);
                Bm.u[1] = *(const uint4*)(vp + 16);
                O[nt] = __builtin_amdgcn_wmma_f32_16x16x32_bf16(false, pA.v, false, Bm.v,
                                                                (short)0, O[nt], false, false);
            }
        }

        wait_async0();      // next chunk's async copies have landed
        __syncthreads();    // ...for every wave in the block
    }

    // ---- normalize and store: out[b, s, h*64 + dh] fp32 ----
    float* ob = out + (size_t)b * SS * DD;
    #pragma unroll
    for (int r = 0; r < 8; ++r) {
        const int row = m0 + r + hihalf * 8;
        const float inv = 1.0f / lrow[r];
        #pragma unroll
        for (int nt = 0; nt < 4; ++nt)
            ob[(size_t)row * DD + h * DH + nt * 16 + l15] = O[nt][r] * inv;
    }
}

// ---------------- launcher ----------------
extern "C" void kernel_launch(void* const* d_in, const int* in_sizes, int n_in,
                              void* d_out, int out_size, void* d_ws, size_t ws_size,
                              hipStream_t stream) {
    const float* X    = (const float*)d_in[0];
    const float* mask = (const float*)d_in[1];
    const float* relp = (const float*)d_in[2];
    const float* rel2 = (const float*)d_in[3];
    const float* Wq   = (const float*)d_in[4];
    const float* bq   = (const float*)d_in[5];
    const float* Wk   = (const float*)d_in[6];
    const float* bk   = (const float*)d_in[7];
    const float* Wv   = (const float*)d_in[8];
    const float* bv   = (const float*)d_in[9];
    float* out = (float*)d_out;

    // workspace carve-up (all bf16 stored as u16), ~28.7 MB total
    unsigned short* wsp = (unsigned short*)d_ws;
    unsigned short* Xb  = wsp;                 wsp += (size_t)BS * DD;   // 4096x768
    unsigned short* Wtq = wsp;                 wsp += (size_t)DD * DD;
    unsigned short* Wtk = wsp;                 wsp += (size_t)DD * DD;
    unsigned short* Wtv = wsp;                 wsp += (size_t)DD * DD;
    unsigned short* Qb  = wsp;                 wsp += (size_t)BB * HH * SS * DH;
    unsigned short* Kb  = wsp;                 wsp += (size_t)BB * HH * SS * DH;
    unsigned short* Vtb = wsp;                 wsp += (size_t)BB * HH * SS * DH;

    const int ncvt = BS * DD;                  // covers both X and W index ranges
    cvt_kernel<<<(ncvt + 255) / 256, 256, 0, stream>>>(X, Wq, Wk, Wv, Xb, Wtq, Wtk, Wtv);

    const int gemmBlocks = (256 * 48) / 8;     // wave-per-tile, 8 waves/block
    qkv_gemm<<<gemmBlocks, 256, 0, stream>>>(Xb, Wtq, bq, Qb, 0);
    qkv_gemm<<<gemmBlocks, 256, 0, stream>>>(Xb, Wtk, bk, Kb, 0);
    qkv_gemm<<<gemmBlocks, 256, 0, stream>>>(Xb, Wtv, bv, Vtb, 1);

    const int attnBlocks = BB * HH * 16;       // 384: (b,h) x 16 row-slabs
    attn_kernel<<<attnBlocks, 256, 0, stream>>>(Qb, Kb, Vtb, relp, rel2, mask, out);
}
